// MambaEncoder_58360015618190
// MI455X (gfx1250) — matrix operand order repaired and checked by reading, not compiled, not run on hardware
//
#include <hip/hip_runtime.h>
#include <hip/hip_bf16.h>
#include <math.h>

// ---------------- problem dimensions (compile-time) ----------------
#define BB   4
#define SS   512
#define CIN  512
#define HH   768
#define II   1536
#define NN   16
#define RR   48
#define KK   4
#define LL   4
#define MM   (BB*SS)          // 2048 token rows
#define EPSF 1e-5f

#define BM    128             // block M-panel (8 waves x 16 rows)
#define KSTEP 32              // K per WMMA step (16x16x32 bf16)

typedef __attribute__((ext_vector_type(16))) __bf16 v16bf;
typedef __attribute__((ext_vector_type(8)))  float  v8f;

// ---- CDNA5 async copy: global -> LDS, 16 bytes per lane (ASYNCcnt) ----
__device__ __forceinline__ void async_copy_b128(unsigned lds_addr, const float* g)
{
    asm volatile("global_load_async_to_lds_b128 %0, %1, off"
                 :: "v"(lds_addr), "v"(g) : "memory");
}
__device__ __forceinline__ void wait_async0()
{
    asm volatile("s_wait_asynccnt 0" ::: "memory");
}
// LDS byte address = low 32 bits of the generic pointer (LDS aperture).
__device__ __forceinline__ unsigned lds_addr_of(const void* p)
{
    return (unsigned)(uintptr_t)p;
}

// ======================================================================
// WMMA GEMM: C[M x Nd] = A[M x K] * B[K x Nd]  (+ epilogue)
//   epi 0: out = acc + (bias ? bias[col] : 0)
//   epi 1: out = softplus(acc + bias[col])
//   epi 2: out = res[row,col] + acc            (residual add)
//
// Block = 256 threads = 8 waves; block computes a 128x16 C panel
// (wave w -> rows bm0+16w .. +15). Per K-step the block stages
//   A-panel 128x32 f32 (16 KB) and B-panel 32x16 f32 (2 KB)
// into double-buffered LDS with global_load_async_to_lds_b128, so the
// async engine fetches block k+1 while WMMA consumes block k.
// B is fetched once per block and shared by all 8 waves.
//
// Fragments follow CDNA5 ISA 7.12.2 (wave32):
//   A (16x32 bf16): lane<16 -> row, K={0..7,16..23}; lane>=16 -> K={8..15,24..31}
//   B (32x16 bf16): lane -> col n0+lane%16, K split as for A
//   C (16x16 f32):  lane<16 rows m0+v, lane>=16 rows m0+8+v, col n0+lane%16
// ======================================================================
__global__ __launch_bounds__(256)
void gemm_wmma_kernel(const float* __restrict__ A, int lda,
                      const float* __restrict__ Bw, int ldb,
                      float* __restrict__ C, int ldc,
                      int M, int Nd, int K,
                      const float* __restrict__ bias,
                      const float* __restrict__ res,
                      int epi)
{
    __shared__ float Abuf[2][BM * KSTEP];   // 2 x 16 KB
    __shared__ float Bbuf[2][KSTEP * 16];   // 2 x 2 KB

    const int tid    = threadIdx.x;
    const int lane   = tid & 31;
    const int wave   = tid >> 5;
    const int tilesN = Nd >> 4;
    const int bm0    = (blockIdx.x / tilesN) * BM;
    const int n0     = (blockIdx.x % tilesN) << 4;

    const int half = lane >> 4;                 // 0 or 1
    const int lrow = wave * 16 + (lane & 15);   // row within the 128-row panel

    // ---- stage one K-block (kb) of A/B panels into LDS buffer bufi ----
    auto stage = [&](int kb, int bufi) {
        if (kb + KSTEP <= K) {
            // async path: A panel = 1024 x 16B chunks, 4 per thread
#pragma unroll
            for (int i = 0; i < 4; ++i) {
                int c  = tid + i * 256;          // chunk 0..1023
                int r  = c >> 3;                 // row 0..127
                int c4 = (c & 7) << 2;           // col 0,4,...,28
                const float* g = A + (size_t)(bm0 + r) * lda + kb + c4;
                async_copy_b128(lds_addr_of(&Abuf[bufi][r * KSTEP + c4]), g);
            }
            // B panel = 128 x 16B chunks, threads 0..127 (wave-uniform split)
            if (tid < 128) {
                int r  = tid >> 2;               // k row 0..31
                int c4 = (tid & 3) << 2;         // col 0,4,8,12
                const float* g = Bw + (size_t)(kb + r) * ldb + n0 + c4;
                async_copy_b128(lds_addr_of(&Bbuf[bufi][r * 16 + c4]), g);
            }
            // L2 hint for the block after next (global_prefetch_b8)
            if (kb + 2 * KSTEP <= K)
                __builtin_prefetch(A + (size_t)(bm0 + (tid >> 1)) * lda + kb + KSTEP, 0, 1);
        } else {
            // K tail (e.g. K=48): guarded synchronous LDS fill, zero-pad
            for (int e = tid; e < BM * KSTEP; e += 256) {
                int r = e >> 5, c = e & 31;
                int k = kb + c;
                Abuf[bufi][e] = (k < K) ? A[(size_t)(bm0 + r) * lda + k] : 0.0f;
            }
            for (int e = tid; e < KSTEP * 16; e += 256) {
                int r = e >> 4, c = e & 15;
                int k = kb + r;
                Bbuf[bufi][e] = (k < K) ? Bw[(size_t)k * ldb + n0 + c] : 0.0f;
            }
        }
    };

    v8f acc;
#pragma unroll
    for (int i = 0; i < 8; ++i) acc[i] = 0.0f;

    // prologue: stage first K-block, then drain ASYNCcnt and barrier
    stage(0, 0);
    wait_async0();
    __syncthreads();

    int cur = 0;
    for (int kb = 0; kb < K; kb += KSTEP) {
        // kick off async fetch of the next K-block into the other buffer
        if (kb + KSTEP < K) stage(kb + KSTEP, cur ^ 1);

        // ---- A fragment from LDS (two contiguous groups of 8 floats) ----
        const float4* pa0 = (const float4*)&Abuf[cur][lrow * KSTEP + half * 8];
        const float4* pa1 = (const float4*)&Abuf[cur][lrow * KSTEP + 16 + half * 8];
        float4 a0 = pa0[0], a1 = pa0[1];
        float4 a2 = pa1[0], a3 = pa1[1];
        v16bf af;
        af[0]=(__bf16)a0.x; af[1]=(__bf16)a0.y; af[2]=(__bf16)a0.z; af[3]=(__bf16)a0.w;
        af[4]=(__bf16)a1.x; af[5]=(__bf16)a1.y; af[6]=(__bf16)a1.z; af[7]=(__bf16)a1.w;
        af[8]=(__bf16)a2.x; af[9]=(__bf16)a2.y; af[10]=(__bf16)a2.z; af[11]=(__bf16)a2.w;
        af[12]=(__bf16)a3.x; af[13]=(__bf16)a3.y; af[14]=(__bf16)a3.z; af[15]=(__bf16)a3.w;

        // ---- B fragment: lane holds col lane%16, K={8h..8h+7, 16+8h..16+8h+7} ----
        const float* pb = &Bbuf[cur][(half * 8) * 16 + (lane & 15)];
        v16bf bfv;
#pragma unroll
        for (int i = 0; i < 8; ++i) {
            bfv[i]     = (__bf16)pb[i * 16];
            bfv[8 + i] = (__bf16)pb[(16 + i) * 16];
        }

        // D = A x B + C  (fp32 accumulate)
        acc = __builtin_amdgcn_wmma_f32_16x16x32_bf16(
                  false, af, false, bfv, (short)0, acc, false, false);

        // drain this iteration's async fetches, then block-wide barrier
        // before the buffers are swapped / overwritten
        wait_async0();
        __syncthreads();
        cur ^= 1;
    }

    // ---- epilogue + store ----
    const int col   = n0 + (lane & 15);
    const int rbase = bm0 + wave * 16 + half * 8;
    const float bcol = (bias != nullptr) ? bias[col] : 0.0f;
#pragma unroll
    for (int v = 0; v < 8; ++v) {
        const int row = rbase + v;
        float o;
        if (epi == 2) {
            o = res[(size_t)row * ldc + col] + acc[v];
        } else {
            o = acc[v] + bcol;
            if (epi == 1) o = (o > 20.0f) ? o : log1pf(__expf(o));  // softplus
        }
        C[(size_t)row * ldc + col] = o;
    }
    (void)M;
}

// ======================================================================
// RMSNorm: one block per row (cols = 768)
// ======================================================================
__global__ void rmsnorm_kernel(const float* __restrict__ x,
                               const float* __restrict__ w,
                               float* __restrict__ out, int cols)
{
    const int row = blockIdx.x;
    const float* xr = x + (size_t)row * cols;
    __shared__ float red[256];
    float s = 0.0f;
    for (int c = threadIdx.x; c < cols; c += blockDim.x) {
        float v = xr[c];
        s += v * v;
    }
    red[threadIdx.x] = s;
    __syncthreads();
    for (int off = 128; off > 0; off >>= 1) {
        if (threadIdx.x < off) red[threadIdx.x] += red[threadIdx.x + off];
        __syncthreads();
    }
    const float rs = rsqrtf(red[0] / (float)cols + EPSF);
    for (int c = threadIdx.x; c < cols; c += blockDim.x)
        out[(size_t)row * cols + c] = xr[c] * rs * w[c];
}

// ======================================================================
// Depthwise causal conv1d (K=4, left pad 3) + bias + SiLU.
// xz is (M, 2I); xc occupies cols [0, I). Output xc (M, I).
// ======================================================================
__global__ void conv_silu_kernel(const float* __restrict__ xz,
                                 const float* __restrict__ cw,
                                 const float* __restrict__ cb,
                                 float* __restrict__ xc, int total)
{
    int idx = blockIdx.x * blockDim.x + threadIdx.x;
    if (idx >= total) return;
    const int i  = idx % II;
    const int bs = idx / II;
    const int s  = bs % SS;
    const int b  = bs / SS;
    float acc = cb[i];
#pragma unroll
    for (int k = 0; k < KK; ++k) {
        int t = s - (KK - 1) + k;
        if (t >= 0)
            acc += cw[i * KK + k] * xz[(size_t)(b * SS + t) * (2 * II) + i];
    }
    const float sig = 1.0f / (1.0f + __expf(-acc));
    xc[idx] = acc * sig;
}

// ======================================================================
// Selective scan, one thread per (b, i): 16-state recurrence over t,
// fused with D skip connection and SiLU(z) gating.
//   dbc row layout: [dt(48) | B(16) | C(16)], stride 80
// ======================================================================
__global__ void scan_kernel(const float* __restrict__ dtbuf,  // (M, I)
                            const float* __restrict__ xcbuf,  // (M, I)
                            const float* __restrict__ dbc,    // (M, 80)
                            const float* __restrict__ xz,     // (M, 2I): z at col I+i
                            const float* __restrict__ A_log,  // (I, N)
                            const float* __restrict__ Dp,     // (I)
                            float* __restrict__ ybuf)         // (M, I)
{
    int idx = blockIdx.x * blockDim.x + threadIdx.x;
    if (idx >= BB * II) return;
    const int b = idx / II;
    const int i = idx % II;

    float Arow[NN], hst[NN];
#pragma unroll
    for (int n = 0; n < NN; ++n) {
        Arow[n] = -__expf(A_log[i * NN + n]);
        hst[n]  = 0.0f;
    }
    const float Di = Dp[i];

    for (int t = 0; t < SS; ++t) {
        const size_t row = (size_t)(b * SS + t);
        const float dt = dtbuf[row * II + i];
        const float xv = xcbuf[row * II + i];
        const float* bc = dbc + row * (RR + 2 * NN);
        const float dtx = dt * xv;
        float y = 0.0f;
#pragma unroll
        for (int n = 0; n < NN; ++n) {
            const float dA = __expf(dt * Arow[n]);
            hst[n] = dA * hst[n] + dtx * bc[RR + n];
            y += hst[n] * bc[RR + NN + n];
        }
        y += xv * Di;
        const float z = xz[row * (2 * II) + II + i];
        y *= z / (1.0f + __expf(-z));            // * silu(z)
        ybuf[row * II + i] = y;
    }
}

// ======================================================================
// Host-side orchestration
// ======================================================================
extern "C" void kernel_launch(void* const* d_in, const int* in_sizes, int n_in,
                              void* d_out, int out_size, void* d_ws, size_t ws_size,
                              hipStream_t stream)
{
    (void)in_sizes; (void)n_in; (void)out_size; (void)ws_size;

    const float* x         = (const float*)d_in[0];   // (B,S,CIN)
    const float* W_in      = (const float*)d_in[1];   // (CIN,H)
    const float* b_in      = (const float*)d_in[2];   // (H)
    const float* norm_w    = (const float*)d_in[3];   // (L,H)
    const float* in_proj_w = (const float*)d_in[4];   // (L,H,2I)
    const float* conv_w    = (const float*)d_in[5];   // (L,I,K)
    const float* conv_b    = (const float*)d_in[6];   // (L,I)
    const float* x_proj_w  = (const float*)d_in[7];   // (L,I,R+2N)
    const float* dt_proj_w = (const float*)d_in[8];   // (L,R,I)
    const float* dt_proj_b = (const float*)d_in[9];   // (L,I)
    const float* A_log     = (const float*)d_in[10];  // (L,I,N)
    const float* Dp        = (const float*)d_in[11];  // (L,I)
    const float* out_proj  = (const float*)d_in[12];  // (L,I,H)
    const float* norm_f_w  = (const float*)d_in[13];  // (H)
    float* outp            = (float*)d_out;           // (B,S,H)

    // ---- workspace carve-up (fp32 elements) ----
    float* wsf    = (float*)d_ws;
    float* hbuf   = wsf;                          // M*H
    float* ubuf   = hbuf   + (size_t)MM * HH;     // M*H
    float* xzbuf  = ubuf   + (size_t)MM * HH;     // M*2I
    float* xcbuf  = xzbuf  + (size_t)MM * 2 * II; // M*I
    float* dbcbuf = xcbuf  + (size_t)MM * II;     // M*80
    float* dtbuf  = dbcbuf + (size_t)MM * (RR + 2 * NN); // M*I
    float* ybuf   = dtbuf  + (size_t)MM * II;     // M*I

    auto gemm = [&](const float* A, int lda, const float* Bw, int ldb,
                    float* C, int ldc, int M, int Nd, int K,
                    const float* bias, const float* res, int epi) {
        int blocks = (M / BM) * (Nd / 16);        // 128x16 panel per block
        gemm_wmma_kernel<<<blocks, 256, 0, stream>>>(A, lda, Bw, ldb, C, ldc,
                                                     M, Nd, K, bias, res, epi);
    };

    // h = x @ W_in + b_in
    gemm(x, CIN, W_in, HH, hbuf, HH, MM, HH, CIN, b_in, nullptr, 0);

    for (int l = 0; l < LL; ++l) {
        const float* nw  = norm_w    + (size_t)l * HH;
        const float* ipw = in_proj_w + (size_t)l * HH * 2 * II;
        const float* cw  = conv_w    + (size_t)l * II * KK;
        const float* cb  = conv_b    + (size_t)l * II;
        const float* xpw = x_proj_w  + (size_t)l * II * (RR + 2 * NN);
        const float* dpw = dt_proj_w + (size_t)l * RR * II;
        const float* dpb = dt_proj_b + (size_t)l * II;
        const float* Al  = A_log     + (size_t)l * II * NN;
        const float* Dl  = Dp        + (size_t)l * II;
        const float* opw = out_proj  + (size_t)l * II * HH;

        // u = rmsnorm(h, nw)
        rmsnorm_kernel<<<MM, 256, 0, stream>>>(hbuf, nw, ubuf, HH);
        // xz = u @ ipw                       (2048 x 768 x 3072)
        gemm(ubuf, HH, ipw, 2 * II, xzbuf, 2 * II, MM, 2 * II, HH,
             nullptr, nullptr, 0);
        // xc = silu(depthwise_conv(xc) + cb)
        conv_silu_kernel<<<(MM * II + 255) / 256, 256, 0, stream>>>(
            xzbuf, cw, cb, xcbuf, MM * II);
        // dbc = xc @ xpw                     (2048 x 1536 x 80)
        gemm(xcbuf, II, xpw, RR + 2 * NN, dbcbuf, RR + 2 * NN, MM,
             RR + 2 * NN, II, nullptr, nullptr, 0);
        // dt = softplus(dbc[:, :48] @ dpw + dpb)   (2048 x 48 x 1536)
        gemm(dbcbuf, RR + 2 * NN, dpw, II, dtbuf, II, MM, II, RR,
             dpb, nullptr, 1);
        // selective scan + D skip + silu(z) gate
        scan_kernel<<<(BB * II + 255) / 256, 256, 0, stream>>>(
            dtbuf, xcbuf, dbcbuf, xzbuf, Al, Dl, ybuf);
        // h = h + y @ opw                    (2048 x 1536 x 768)
        gemm(ybuf, II, opw, HH, hbuf, HH, MM, HH, II, nullptr, hbuf, 2);
    }

    // out = rmsnorm(h, norm_f_w)
    rmsnorm_kernel<<<MM, 256, 0, stream>>>(hbuf, norm_f_w, outp, HH);
}
